// ResidualConvBlock_24103356465408
// MI455X (gfx1250) — compile-verified, matrix operand order checked by reading, not run on hardware
//
#include <hip/hip_runtime.h>

// Problem constants (match reference)
#define BB 4
#define CC 128
#define NN 40962
#define KK 9
#define TM 32                      // vertices per workgroup tile
#define TILES32 ((NN + TM - 1) / TM)   // 1281 vertex tiles per batch
#define TILES16 ((NN + 15) / 16)       // 2561 (prep_features tiles)
#define KTOT  (CC * KK)            // 1152 gathered K dimension
#define KSTEPS (KTOT / 32)         // 36 WMMA K-steps (bf16 k=32)

typedef __attribute__((ext_vector_type(16))) __bf16 v16bf;
typedef __attribute__((ext_vector_type(8)))  float  v8f;

union V16U { v16bf v; uint4 q[2]; };

__device__ __forceinline__ __bf16 f2bf(float f) {
  unsigned u = __builtin_bit_cast(unsigned, f);
  unsigned r = (u + 0x7FFFu + ((u >> 16) & 1u)) >> 16;   // RNE
  unsigned short h = (unsigned short)r;
  return __builtin_bit_cast(__bf16, h);
}

// Zero the pad row (vertex index N) of both staged feature buffers.
__global__ void zero_pad_rows(__bf16* __restrict__ fbuf, __bf16* __restrict__ gbuf) {
  const int b = blockIdx.x;
  const int c = threadIdx.x;                     // 128 threads
  const size_t off = ((size_t)b * (NN + 1) + NN) * CC + c;
  const __bf16 z = __builtin_bit_cast(__bf16, (unsigned short)0);
  fbuf[off] = z;
  gbuf[off] = z;
}

// Convert W[o][c][k] fp32 -> bf16 in WMMA B-fragment order:
// wb[((s*8+g)*32 + lane)*16 + t]; K-step s covers neighbor k=s/4, c in [(s%4)*32,+32).
// Lane packing mirrors the 16-bit A-matrix layout (lanes 0-15: K{0..7,16..23}).
__global__ void prep_weights(const float* __restrict__ W1, const float* __restrict__ W2,
                             __bf16* __restrict__ wb1, __bf16* __restrict__ wb2) {
  int tid = blockIdx.x * blockDim.x + threadIdx.x;   // < 2*147456
  int layer = tid / (KSTEPS * 4096);
  int e = tid - layer * (KSTEPS * 4096);
  int t16 = e & 15;
  int l   = (e >> 4) & 31;
  int g   = (e >> 9) & 7;
  int s   = e >> 12;
  int r = (t16 < 8 ? t16 : t16 + 8) + (l < 16 ? 0 : 8);   // K-row within 32-chunk
  int j = l & 15;                                         // output-channel column
  int kn = s >> 2;
  int c  = ((s & 3) << 5) + r;
  int o  = (g << 4) + j;
  const float* W = layer ? W2 : W1;
  __bf16* wb = layer ? wb2 : wb1;
  wb[e] = f2bf(W[((size_t)o * CC + c) * KK + kn]);
}

// f = relu(x) as bf16, transposed [B,C,N] -> vertex-major [B][N+1][128].
__global__ void __launch_bounds__(256)
prep_features(const float* __restrict__ x, __bf16* __restrict__ fbuf) {
  __shared__ __bf16 sh[16 * CC];
  const int t = threadIdx.x;
  const int blk = blockIdx.x;
  const int b = blk / TILES16;
  const int n0 = (blk - b * TILES16) * 16;
  const int nl = t & 15, crow = t >> 4;
#pragma unroll
  for (int i = 0; i < 8; ++i) {
    int c = (i << 4) + crow;
    int n = n0 + nl;
    float v = (n < NN) ? x[((size_t)b * CC + c) * NN + n] : 0.0f;
    v = v > 0.0f ? v : 0.0f;
    sh[nl * CC + c] = f2bf(v);
  }
  __syncthreads();
  const int m = t >> 4, c0 = (t & 15) << 3;
  if (n0 + m < NN)
    *(uint4*)(fbuf + ((size_t)b * (NN + 1) + n0 + m) * CC + c0) =
        *(const uint4*)(sh + m * CC + c0);
}

// One sphconv layer. 256 threads = 8 waves; wave g owns output channels [16g,16g+16)
// for all 32 vertices of the tile (two 16x16 D tiles, shared B fragment).
// FINAL=false: out = relu(conv + b) -> bf16 vertex-major gout.
// FINAL=true : out = conv + b + x   -> fp32 channel-major d_out.
template<bool FINAL>
__global__ void __launch_bounds__(256)
sphconv_layer(const __bf16* __restrict__ fin,
              const __bf16* __restrict__ wb,
              const float* __restrict__ bias,
              const long long* __restrict__ nmap,
              const float* __restrict__ xres,
              __bf16* __restrict__ gout,
              float* __restrict__ out) {
  __shared__ __align__(16) __bf16 Atile[TM * KTOT];  // 72 KB; reused as epilogue staging
  __shared__ int idx[TM * KK];

  const int t = threadIdx.x;
  const int blk = blockIdx.x;
  const int b = blk / TILES32;
  const int n0 = (blk - b * TILES32) * TM;

  for (int ii = t; ii < TM * KK; ii += 256) {       // 288 neighbor indices
    int m = ii / KK, kk2 = ii - m * KK;
    int n = n0 + m;
    idx[ii] = (n < NN) ? (int)nmap[(size_t)n * KK + kk2] : NN;  // pad row for ragged tail
  }
  __syncthreads();

  // Cooperative gather: 288 neighbor segments x 256B each, 16B per thread-slot.
  const __bf16* fb = fin + (size_t)b * (NN + 1) * CC;
#pragma unroll
  for (int i = 0; i < 18; ++i) {
    int slot = i * 256 + t;
    int seg  = slot >> 4;               // 0..287
    int off8 = (slot & 15) << 3;        // element offset within 128-wide segment
    int m = seg / KK, kk2 = seg - m * KK;
    *(uint4*)(Atile + m * KTOT + kk2 * CC + off8) =
        *(const uint4*)(fb + (size_t)idx[seg] * CC + off8);
  }
  __syncthreads();

  const int wave = t >> 5, lane = t & 31;
  const int mrow = lane & 15, half = lane >> 4;
  const __bf16* arow0 = Atile + mrow * KTOT + (half << 3);
  const __bf16* arow1 = arow0 + 16 * KTOT;
  const uint4* wrow = (const uint4*)wb + (((size_t)wave * 32 + lane) << 1);

  v8f acc0 = {}, acc1 = {};
#pragma unroll 2
  for (int s = 0; s < KSTEPS; ++s) {
    V16U a0, a1, w;
    const __bf16* ap0 = arow0 + s * 32;
    a0.q[0] = *(const uint4*)(ap0);        // K 0..7   (lanes 0-15) / 8..15  (16-31)
    a0.q[1] = *(const uint4*)(ap0 + 16);   // K 16..23 (lanes 0-15) / 24..31 (16-31)
    const __bf16* ap1 = arow1 + s * 32;
    a1.q[0] = *(const uint4*)(ap1);
    a1.q[1] = *(const uint4*)(ap1 + 16);
    const uint4* wp = wrow + ((size_t)s << 9);   // +s*4096 bf16 elems
    w.q[0] = wp[0];
    w.q[1] = wp[1];
    // Two independent accumulation chains sharing one B fragment.
    acc0 = __builtin_amdgcn_wmma_f32_16x16x32_bf16(
        false, a0.v, false, w.v, (short)0, acc0, false, false);
    acc1 = __builtin_amdgcn_wmma_f32_16x16x32_bf16(
        false, a1.v, false, w.v, (short)0, acc1, false, false);
  }

  const float bi = bias[(wave << 4) + mrow];
  __syncthreads();   // all waves done reading Atile; safe to reuse for epilogue staging

  if (!FINAL) {
    __bf16* outB = (__bf16*)Atile;           // 32 x 128 bf16 = 8 KB
#pragma unroll
    for (int r = 0; r < 8; ++r) {
      int m0 = r + (half << 3);              // D layout: lanes16-31 hold M=8+r
      float v0 = acc0[r] + bi;
      v0 = v0 > 0.0f ? v0 : 0.0f;
      outB[m0 * CC + (wave << 4) + mrow] = f2bf(v0);
      float v1 = acc1[r] + bi;
      v1 = v1 > 0.0f ? v1 : 0.0f;
      outB[(m0 + 16) * CC + (wave << 4) + mrow] = f2bf(v1);
    }
    __syncthreads();
#pragma unroll
    for (int j = 0; j < 2; ++j) {
      int tt = j * 256 + t;
      int m = tt >> 4, c0 = (tt & 15) << 3;
      if (n0 + m < NN)
        *(uint4*)(gout + ((size_t)b * (NN + 1) + n0 + m) * CC + c0) =
            *(const uint4*)(outB + m * CC + c0);
    }
  } else {
    float* outF = (float*)Atile;             // 32 x 128 f32 = 16 KB
#pragma unroll
    for (int r = 0; r < 8; ++r) {
      int m0 = r + (half << 3);
      outF[m0 * CC + (wave << 4) + mrow] = acc0[r] + bi;
      outF[(m0 + 16) * CC + (wave << 4) + mrow] = acc1[r] + bi;
    }
    __syncthreads();
#pragma unroll
    for (int i = 0; i < 16; ++i) {
      int flat = i * 256 + t;
      int o = flat >> 5, m = flat & 31;      // 32 lanes cover 32 consecutive n: coalesced
      int n = n0 + m;
      if (n < NN) {
        size_t a2 = ((size_t)b * CC + o) * NN + n;
        out[a2] = outF[m * CC + o] + xres[a2];
      }
    }
  }
}

extern "C" void kernel_launch(void* const* d_in, const int* in_sizes, int n_in,
                              void* d_out, int out_size, void* d_ws, size_t ws_size,
                              hipStream_t stream) {
  const float*     x    = (const float*)d_in[0];
  const long long* nmap = (const long long*)d_in[1];
  const float*     W1   = (const float*)d_in[2];
  const float*     b1   = (const float*)d_in[3];
  const float*     W2   = (const float*)d_in[4];
  const float*     b2   = (const float*)d_in[5];
  float* out = (float*)d_out;

  // Workspace: fbuf (bf16 relu(x), vertex-major) | gbuf (bf16 relu(h1)) | wb1 | wb2
  char* ws = (char*)d_ws;
  const size_t F = (size_t)BB * (NN + 1) * CC * sizeof(__bf16);   // ~40 MB each
  __bf16* fbuf = (__bf16*)ws;
  __bf16* gbuf = (__bf16*)(ws + F);
  __bf16* wb1  = (__bf16*)(ws + 2 * F);
  __bf16* wb2  = wb1 + (size_t)KSTEPS * 4096;

  zero_pad_rows<<<BB, CC, 0, stream>>>(fbuf, gbuf);
  prep_weights<<<(2 * KSTEPS * 4096) / 256, 256, 0, stream>>>(W1, W2, wb1, wb2);
  prep_features<<<BB * TILES16, 256, 0, stream>>>(x, fbuf);
  sphconv_layer<false><<<BB * TILES32, 256, 0, stream>>>(fbuf, wb1, b1, nmap, nullptr, gbuf, nullptr);
  sphconv_layer<true><<<BB * TILES32, 256, 0, stream>>>(gbuf, wb2, b2, nmap, x, nullptr, out);
}
